// TransformerBlock_34394098106968
// MI455X (gfx1250) — matrix operand order repaired and checked by reading, not compiled, not run on hardware
//
#include <hip/hip_runtime.h>
#include <hip/hip_bf16.h>
#include <stdint.h>

typedef __bf16 bf16;
typedef __attribute__((ext_vector_type(16))) __bf16 v16bf;
typedef __attribute__((ext_vector_type(8)))  float  v8f;
typedef __attribute__((ext_vector_type(4)))  unsigned int v4u;
typedef __attribute__((ext_vector_type(8)))  int v8i_;
typedef __attribute__((ext_vector_type(4)))  int v4i_;

static constexpr int kD  = 1024;
static constexpr int kT  = 2048;
static constexpr int kB  = 2;
static constexpr int kH  = 16;
static constexpr int kHD = 64;
static constexpr int kF  = 4096;
static constexpr int kBT = kB * kT;          // 4096 rows
static constexpr float kEPS = 1e-5f;
static constexpr float kNEG = -3.0e38f;

union Frag32 { uint4 u[2]; v16bf v; };

__device__ __forceinline__ v16bf load_frag(const bf16* p0, const bf16* p1) {
  Frag32 f;
  f.u[0] = *reinterpret_cast<const uint4*>(p0);
  f.u[1] = *reinterpret_cast<const uint4*>(p1);
  return f.v;
}

__device__ __forceinline__ v8f zero8() {
  v8f z = {0.f, 0.f, 0.f, 0.f, 0.f, 0.f, 0.f, 0.f};
  return z;
}

__device__ __forceinline__ v8f wmma_bf16(v16bf a, v16bf b, v8f c) {
  // D = A(16x32 bf16) * B(32x16 bf16) + C(16x16 f32)
  return __builtin_amdgcn_wmma_f32_16x16x32_bf16(
      false, a, false, b, (short)0, c, false, false);
}

__device__ __forceinline__ float red_max16(float v) {
  v = fmaxf(v, __shfl_xor(v, 1, 32));
  v = fmaxf(v, __shfl_xor(v, 2, 32));
  v = fmaxf(v, __shfl_xor(v, 4, 32));
  v = fmaxf(v, __shfl_xor(v, 8, 32));
  return v;
}
__device__ __forceinline__ float red_sum16(float v) {
  v += __shfl_xor(v, 1, 32);
  v += __shfl_xor(v, 2, 32);
  v += __shfl_xor(v, 4, 32);
  v += __shfl_xor(v, 8, 32);
  return v;
}

__device__ __forceinline__ float gelu_exact(float x) {
  return 0.5f * x * (1.0f + erff(x * 0.70710678118654752f));
}

// --------------------------------------------------------- TDM 2D tile load
// tensor_load_to_lds of a [rows x 64] bf16 tile, global row stride
// `stride_elems`, into LDS with a 4-DWORD pad after every 32 DWORDs
// (i.e. rows of 64 bf16 padded to 72 -> matches the ST=72 LDS layout).
// D# bitfields per CDNA5 ISA.
__device__ __forceinline__ void tdm_load_tile_2d(uint32_t lds_addr,
                                                 const void* gaddr,
                                                 uint32_t rows,
                                                 uint32_t stride_elems) {
  const uint64_t ga = (uint64_t)(uintptr_t)gaddr;
  v4u g0;
  g0[0] = 1u;                                        // count=1, user descriptor
  g0[1] = lds_addr;                                  // [63:32] lds_addr
  g0[2] = (uint32_t)(ga & 0xFFFFFFFFu);              // [95:64] global_addr lo
  g0[3] = (uint32_t)((ga >> 32) & 0x1FFFFFFu)        // [120:96] global_addr hi
        | (2u << 30);                                // [127:126] type=2 (image)
  const uint32_t td0 = stride_elems;                 // tensor_dim0 (row length)
  const uint32_t td1 = rows;                         // tensor_dim1
  v8i_ g1;
  g1[0] = (int)((1u << 16)                           // data_size = 2 bytes
              | (1u << 20)                           // pad_enable
              | (4u << 22)                           // pad_interval: 32 DWORDs
              | (3u << 25));                         // pad_amount: 4 DWORDs
  g1[1] = (int)((td0 & 0xFFFFu) << 16);              // tensor_dim0[15:0]
  g1[2] = (int)(((td0 >> 16) & 0xFFFFu)              // tensor_dim0[31:16]
              | ((td1 & 0xFFFFu) << 16));            // tensor_dim1[15:0]
  g1[3] = (int)(((td1 >> 16) & 0xFFFFu)              // tensor_dim1[31:16]
              | (64u << 16));                        // tile_dim0 = 64 elems
  g1[4] = (int)rows;                                 // tile_dim1 = rows, dim2=0
  g1[5] = (int)stride_elems;                         // tensor_dim0_stride lo
  g1[6] = 0;                                         // stride0 hi | stride1 lo
  g1[7] = 0;
  v4i_ z4 = {0, 0, 0, 0};
  v8i_ z8 = {0, 0, 0, 0, 0, 0, 0, 0};
  __builtin_amdgcn_tensor_load_to_lds(g0, g1, z4, z4, z8, 0);
  __asm__ __volatile__("" ::: "memory");
}

// ---------------------------------------------------------------- converts
__global__ void f32_to_bf16_kernel(const float* __restrict__ in,
                                   bf16* __restrict__ out, int n) {
  int i = blockIdx.x * blockDim.x + threadIdx.x;
  if (i < n) out[i] = (bf16)in[i];
}

// ---------------------------------------------------------------- layernorm
__global__ __launch_bounds__(256)
void layernorm_bf16_kernel(const float* __restrict__ x,
                           const float* __restrict__ g,
                           const float* __restrict__ b,
                           bf16* __restrict__ out) {
  __shared__ float red_s[8];
  __shared__ float red_q[8];
  const int row = blockIdx.x;
  const int tid = threadIdx.x;
  const float* xr = x + (size_t)row * kD;
  float4 v = *reinterpret_cast<const float4*>(xr + tid * 4);
  float s  = v.x + v.y + v.z + v.w;
  float s2 = v.x * v.x + v.y * v.y + v.z * v.z + v.w * v.w;
  for (int m = 16; m >= 1; m >>= 1) {
    s  += __shfl_xor(s,  m, 32);
    s2 += __shfl_xor(s2, m, 32);
  }
  const int wave = tid >> 5, lane = tid & 31;
  if (lane == 0) { red_s[wave] = s; red_q[wave] = s2; }
  __syncthreads();
  s = 0.f; s2 = 0.f;
#pragma unroll
  for (int i = 0; i < 8; i++) { s += red_s[i]; s2 += red_q[i]; }
  const float mu   = s * (1.0f / kD);
  const float var  = s2 * (1.0f / kD) - mu * mu;
  const float rstd = rsqrtf(var + kEPS);
  bf16* orow = out + (size_t)row * kD;
  const int c = tid * 4;
  orow[c + 0] = (bf16)((v.x - mu) * rstd * g[c + 0] + b[c + 0]);
  orow[c + 1] = (bf16)((v.y - mu) * rstd * g[c + 1] + b[c + 1]);
  orow[c + 2] = (bf16)((v.z - mu) * rstd * g[c + 2] + b[c + 2]);
  orow[c + 3] = (bf16)((v.w - mu) * rstd * g[c + 3] + b[c + 3]);
}

// ---------------------------------------------------------------- WMMA GEMM
// C[M,N] = act(A_bf16[M,K] @ W_bf16[K,N] + bias) (+ resid) -> f32 or bf16
// Block tile 128x128, BK=64, 256 threads (8 waves), wave tile 32x64.
// A-tiles move via the Tensor Data Mover into a double-buffered LDS stage
// (hardware row padding to ST=72), with the next DMA in flight behind the
// WMMAs (TENSORcnt). W-tiles stage through VGPRs (transposed on commit).
template <bool GELU_ACT, bool RESID, bool OUT_F32>
__global__ __launch_bounds__(256)
void gemm_bf16_kernel(const bf16* __restrict__ A, const bf16* __restrict__ W,
                      const float* __restrict__ bias,
                      const float* __restrict__ resid,
                      float* __restrict__ outF, bf16* __restrict__ outB,
                      int M, int N, int K) {
  constexpr int BM = 128, BN = 128, BK = 64, ST = BK + 8;
  __shared__ __align__(16) bf16 lds_a[2][BM][ST];  // TDM double buffer [m][k]
  __shared__ __align__(16) bf16 lds_b[BN][ST];     // N-major [n][k]
  const int m0 = blockIdx.y * BM;
  const int n0 = blockIdx.x * BN;
  const int t = threadIdx.x;
  const int wave = t >> 5, lane = t & 31;
  const int half = lane >> 4, l16 = lane & 15;
  const int wm = (wave >> 1) * 32;
  const int wn = (wave & 1) * 64;

  uint4 rb[4];  // register staging for one W K-stage

  auto issue_tdm = [&](int kb, int buf) {
    if (wave == 0)
      tdm_load_tile_2d((uint32_t)(uintptr_t)&lds_a[buf][0][0],
                       &A[(size_t)m0 * K + kb], BM, (uint32_t)K);
  };
  auto load_w = [&](int kb) {
#pragma unroll
    for (int i = 0; i < 4; i++) {            // W: 64x128 = 1024 uint4
      int idx = t + i * 256;
      int kk = idx >> 4, n8 = (idx & 15) * 8;
      rb[i] = *reinterpret_cast<const uint4*>(&W[(size_t)(kb + kk) * N + n0 + n8]);
    }
  };
  auto store_w = [&]() {
#pragma unroll
    for (int i = 0; i < 4; i++) {            // transpose W chunk: [k][n]->[n][k]
      int idx = t + i * 256;
      int kk = idx >> 4, n8 = (idx & 15) * 8;
      const bf16* e = reinterpret_cast<const bf16*>(&rb[i]);
#pragma unroll
      for (int j = 0; j < 8; j++) lds_b[n8 + j][kk] = e[j];
    }
  };

  v8f acc[2][4];
#pragma unroll
  for (int i = 0; i < 2; i++)
#pragma unroll
    for (int j = 0; j < 4; j++) acc[i][j] = zero8();

  issue_tdm(0, 0);     // first A tile DMA in flight
  load_w(0);
  store_w();

  int buf = 0;
  for (int kb = 0; kb < K; kb += BK, buf ^= 1) {
    const bool more = (kb + BK) < K;
    if (more) issue_tdm(kb + BK, buf ^ 1);    // next DMA behind the WMMAs
    if (wave == 0) {
      if (more) __builtin_amdgcn_s_wait_tensorcnt(1);
      else      __builtin_amdgcn_s_wait_tensorcnt(0);
    }
    __syncthreads();                          // A[buf] + lds_b stage visible
    if (more) load_w(kb + BK);                // W HBM loads fly during WMMAs
#pragma unroll
    for (int ks = 0; ks < 2; ks++) {
      v16bf af[2], bfr[4];
#pragma unroll
      for (int i = 0; i < 2; i++) {
        int m = wm + i * 16 + l16;
        int c0 = ks * 32 + half * 8;          // A layout: K chunks [c0, c0+8) and [c0+16, c0+24)
        af[i] = load_frag(&lds_a[buf][m][c0], &lds_a[buf][m][c0 + 16]);
      }
#pragma unroll
      for (int j = 0; j < 4; j++) {
        int n = wn + j * 16 + l16;
        int k0 = ks * 32 + half * 8;          // B layout: K chunks [k0, k0+8) and [k0+16, k0+24)
        bfr[j] = load_frag(&lds_b[n][k0], &lds_b[n][k0 + 16]);
      }
#pragma unroll
      for (int i = 0; i < 2; i++)
#pragma unroll
        for (int j = 0; j < 4; j++)
          acc[i][j] = wmma_bf16(af[i], bfr[j], acc[i][j]);
    }
    __syncthreads();                          // all LDS reads complete
    if (more) store_w();                      // commit next W stage
  }

  // --- epilogue: C element r <-> (m = r + half*8, n = l16)
#pragma unroll
  for (int i = 0; i < 2; i++)
#pragma unroll
    for (int j = 0; j < 4; j++) {
      int col = n0 + wn + j * 16 + l16;
      float bcol = bias[col];
#pragma unroll
      for (int r = 0; r < 8; r++) {
        int row = m0 + wm + i * 16 + r + half * 8;
        float val = acc[i][j][r] + bcol;
        if (GELU_ACT) val = gelu_exact(val);
        if (RESID)    val += resid[(size_t)row * N + col];
        if (OUT_F32)  outF[(size_t)row * N + col] = val;
        else          outB[(size_t)row * N + col] = (bf16)val;
      }
    }
}

// ---------------------------------------------------------------- attention
// Flash-style causal attention. Block = (b, h, 64-row q tile), 4 waves,
// each wave owns 16 q rows and the full HD=64 output.
__global__ __launch_bounds__(128)
void attn_kernel(const bf16* __restrict__ q, const bf16* __restrict__ k,
                 const bf16* __restrict__ v, bf16* __restrict__ ctx) {
  constexpr int ST = kHD + 8;
  __shared__ __align__(16) bf16 lds_k [64][ST];      // [kj][d]  (B for S)
  __shared__ __align__(16) bf16 lds_vt[64][ST];      // [d][kj]  (B for ctx)
  __shared__ __align__(16) bf16 lds_p [4][16][ST];   // per-wave P staging
  const int blk = blockIdx.x;
  const int qt = blk % (kT / 64);
  const int h  = (blk / (kT / 64)) % kH;
  const int bz = blk / ((kT / 64) * kH);
  const int t = threadIdx.x;
  const int wave = t >> 5, lane = t & 31;
  const int half = lane >> 4, l16 = lane & 15;
  const int q0 = qt * 64 + wave * 16;                // this wave's first q row
  const float scale = 0.125f;                        // 1/sqrt(64)

  // Q A-fragments (held for whole kernel)
  v16bf aq[2];
  {
    size_t base = ((size_t)(bz * kT) + q0 + l16) * kD + h * kHD;
#pragma unroll
    for (int ks = 0; ks < 2; ks++) {
      int c0 = ks * 32 + half * 8;
      aq[ks] = load_frag(&q[base + c0], &q[base + c0 + 16]);
    }
  }

  v8f acc[4];
#pragma unroll
  for (int dt = 0; dt < 4; dt++) acc[dt] = zero8();
  float mrow[8], lrow[8];
#pragma unroll
  for (int r = 0; r < 8; r++) { mrow[r] = kNEG; lrow[r] = 0.f; }

  for (int kt = 0; kt <= qt; kt++) {
    const int k0 = kt * 64;
    __syncthreads();
    // stage K tile row-major and V tile transposed (128 threads, 4 uint4 each)
#pragma unroll
    for (int i = 0; i < 4; i++) {
      int idx = t + i * 128;                // 0..511
      int kr = idx >> 3, d8 = (idx & 7) * 8;
      size_t g = ((size_t)(bz * kT) + k0 + kr) * kD + h * kHD + d8;
      *reinterpret_cast<uint4*>(&lds_k[kr][d8]) =
          *reinterpret_cast<const uint4*>(&k[g]);
      uint4 u = *reinterpret_cast<const uint4*>(&v[g]);
      const bf16* e = reinterpret_cast<const bf16*>(&u);
#pragma unroll
      for (int j = 0; j < 8; j++) lds_vt[d8 + j][kr] = e[j];
    }
    __syncthreads();

    // S = Q @ K^T   (4 tiles of 16 kj)
    v8f s[4];
#pragma unroll
    for (int nt = 0; nt < 4; nt++) {
      v8f c = zero8();
#pragma unroll
      for (int ks = 0; ks < 2; ks++) {
        int kj = nt * 16 + l16;
        int d0 = ks * 32 + half * 8;
        v16bf bk = load_frag(&lds_k[kj][d0], &lds_k[kj][d0 + 16]);
        c = wmma_bf16(aq[ks], bk, c);
      }
      s[nt] = c;
    }

    const bool maskTile = (kt == qt);
    // online softmax update
#pragma unroll
    for (int r = 0; r < 8; r++) {
      const int qg = q0 + r + half * 8;
      float mv = kNEG;
#pragma unroll
      for (int nt = 0; nt < 4; nt++) {
        float xv = s[nt][r] * scale;
        if (maskTile) {
          int kg = k0 + nt * 16 + l16;
          if (kg > qg) xv = kNEG;
        }
        s[nt][r] = xv;
        mv = fmaxf(mv, xv);
      }
      mv = red_max16(mv);
      const float mnew = fmaxf(mrow[r], mv);
      const float alpha = expf(mrow[r] - mnew);
      mrow[r] = mnew;
      lrow[r] *= alpha;
#pragma unroll
      for (int dt = 0; dt < 4; dt++) acc[dt][r] = acc[dt][r] * alpha;
      float ps = 0.f;
#pragma unroll
      for (int nt = 0; nt < 4; nt++) {
        float p = expf(s[nt][r] - mnew);
        ps += p;
        lds_p[wave][r + half * 8][nt * 16 + l16] = (bf16)p;  // C -> staging
      }
      lrow[r] += red_sum16(ps);
    }

    // ctx += P @ V   (A from staged P, B from V^T tile)
#pragma unroll
    for (int kjs = 0; kjs < 2; kjs++) {
      int c0 = kjs * 32 + half * 8;
      v16bf pa = load_frag(&lds_p[wave][l16][c0], &lds_p[wave][l16][c0 + 16]);
#pragma unroll
      for (int dt = 0; dt < 4; dt++) {
        int d = dt * 16 + l16;
        int kk0 = kjs * 32 + half * 8;
        v16bf bv_ = load_frag(&lds_vt[d][kk0], &lds_vt[d][kk0 + 16]);
        acc[dt] = wmma_bf16(pa, bv_, acc[dt]);
      }
    }
  }

  // normalize + write ctx back in [B,T,H,HD] bf16 layout
#pragma unroll
  for (int dt = 0; dt < 4; dt++) {
#pragma unroll
    for (int r = 0; r < 8; r++) {
      int qg = q0 + r + half * 8;
      int d = dt * 16 + l16;
      float o = acc[dt][r] / lrow[r];
      ctx[((size_t)(bz * kT) + qg) * kD + h * kHD + d] = (bf16)o;
    }
  }
}

// ---------------------------------------------------------------- launch
extern "C" void kernel_launch(void* const* d_in, const int* in_sizes, int n_in,
                              void* d_out, int out_size, void* d_ws, size_t ws_size,
                              hipStream_t stream) {
  (void)in_sizes; (void)n_in; (void)out_size; (void)ws_size;
  const float* x    = (const float*)d_in[0];
  const float* ln1g = (const float*)d_in[1];
  const float* ln1b = (const float*)d_in[2];
  const float* Wq   = (const float*)d_in[3];
  const float* bq   = (const float*)d_in[4];
  const float* Wk   = (const float*)d_in[5];
  const float* bk   = (const float*)d_in[6];
  const float* Wv   = (const float*)d_in[7];
  const float* bv   = (const float*)d_in[8];
  const float* Wo   = (const float*)d_in[9];
  const float* bo   = (const float*)d_in[10];
  const float* ln2g = (const float*)d_in[11];
  const float* ln2b = (const float*)d_in[12];
  const float* W1   = (const float*)d_in[13];
  const float* b1   = (const float*)d_in[14];
  const float* W2   = (const float*)d_in[15];
  const float* b2   = (const float*)d_in[16];

  uint8_t* ws = (uint8_t*)d_ws;
  size_t off = 0;
  auto carve = [&](size_t bytes) -> void* {
    void* p = ws + off;
    off += (bytes + 255) & ~(size_t)255;
    return p;
  };
  const size_t Dsq = (size_t)kD * kD;
  bf16*  wq_b  = (bf16*)carve(Dsq * 2);
  bf16*  wk_b  = (bf16*)carve(Dsq * 2);
  bf16*  wv_b  = (bf16*)carve(Dsq * 2);
  bf16*  wo_b  = (bf16*)carve(Dsq * 2);
  bf16*  w1_b  = (bf16*)carve((size_t)kD * kF * 2);
  bf16*  w2_b  = (bf16*)carve((size_t)kF * kD * 2);
  bf16*  h1    = (bf16*)carve((size_t)kBT * kD * 2);
  bf16*  q_b   = (bf16*)carve((size_t)kBT * kD * 2);
  bf16*  k_b   = (bf16*)carve((size_t)kBT * kD * 2);
  bf16*  v_b   = (bf16*)carve((size_t)kBT * kD * 2);
  bf16*  ctx_b = (bf16*)carve((size_t)kBT * kD * 2);
  float* x1    = (float*)carve((size_t)kBT * kD * 4);
  bf16*  h2    = (bf16*)carve((size_t)kBT * kD * 2);
  bf16*  mh    = (bf16*)carve((size_t)kBT * kF * 2);

  auto cvt = [&](const float* src, bf16* dst, size_t n) {
    f32_to_bf16_kernel<<<dim3((unsigned)((n + 255) / 256)), dim3(256), 0, stream>>>(
        src, dst, (int)n);
  };
  cvt(Wq, wq_b, Dsq);
  cvt(Wk, wk_b, Dsq);
  cvt(Wv, wv_b, Dsq);
  cvt(Wo, wo_b, Dsq);
  cvt(W1, w1_b, (size_t)kD * kF);
  cvt(W2, w2_b, (size_t)kF * kD);

  // LN1: x -> h1 (bf16)
  layernorm_bf16_kernel<<<dim3(kBT), dim3(256), 0, stream>>>(x, ln1g, ln1b, h1);

  // QKV projections
  gemm_bf16_kernel<false, false, false><<<dim3(kD / 128, kBT / 128), dim3(256), 0, stream>>>(
      h1, wq_b, bq, nullptr, nullptr, q_b, kBT, kD, kD);
  gemm_bf16_kernel<false, false, false><<<dim3(kD / 128, kBT / 128), dim3(256), 0, stream>>>(
      h1, wk_b, bk, nullptr, nullptr, k_b, kBT, kD, kD);
  gemm_bf16_kernel<false, false, false><<<dim3(kD / 128, kBT / 128), dim3(256), 0, stream>>>(
      h1, wv_b, bv, nullptr, nullptr, v_b, kBT, kD, kD);

  // causal flash attention
  attn_kernel<<<dim3(kB * kH * (kT / 64)), dim3(128), 0, stream>>>(q_b, k_b, v_b, ctx_b);

  // out-proj + residual -> x1 (f32)
  gemm_bf16_kernel<false, true, true><<<dim3(kD / 128, kBT / 128), dim3(256), 0, stream>>>(
      ctx_b, wo_b, bo, x, x1, nullptr, kBT, kD, kD);

  // LN2: x1 -> h2 (bf16)
  layernorm_bf16_kernel<<<dim3(kBT), dim3(256), 0, stream>>>(x1, ln2g, ln2b, h2);

  // MLP up + exact GELU -> mh (bf16)
  gemm_bf16_kernel<true, false, false><<<dim3(kF / 128, kBT / 128), dim3(256), 0, stream>>>(
      h2, w1_b, b1, nullptr, nullptr, mh, kBT, kF, kD);

  // MLP down + residual -> d_out (f32)
  gemm_bf16_kernel<false, true, true><<<dim3(kD / 128, kBT / 128), dim3(256), 0, stream>>>(
      mh, w2_b, b2, x1, (float*)d_out, nullptr, kBT, kD, kF);
}
